// Rewriter_9199819948219
// MI455X (gfx1250) — compile-verified, matrix operand order checked by reading, not run on hardware
//
#include <hip/hip_runtime.h>
#include <hip/hip_bf16.h>

typedef __attribute__((ext_vector_type(16))) _Float16 v16h;
typedef __attribute__((ext_vector_type(8)))  _Float16 v8h;
typedef __attribute__((ext_vector_type(8)))  float    v8f;

#define BN 32
#define T_ENC 256
#define EMB 256
#define ENC_H 256
#define PROJ 128
#define HEADS 4
#define HD 32
#define DEC_H 256
#define DEC_O 128
#define VOCAB 30
#define MAX_STEPS 600
#define PAD_TOK 29
#define SOS_TOK 0
#define ATT_SCALE 0.17677669529663687f  /* 1/sqrt(32) */

__device__ __forceinline__ float sigm(float x) { return 1.f / (1.f + __expf(-x)); }

// -------------------------- WMMA fragment loader ---------------------------
// CDNA5 ISA 7.12.2 layout for 16-bit A/B 16x32 (K-major per lane):
// lane L holds M(or N)=L%16; elements 0..7 cover K = 8*(L/16)+e,
// elements 8..15 cover K = 16 + 8*(L/16)+(e-8).
// f16 source: two contiguous 16-byte vector loads per lane.
// Requires ld % 8 == 0 and 16-byte-aligned base (all our operands qualify).
__device__ __forceinline__ v16h frag_f16(const _Float16* P, int ld, int k0) {
  const int lane = threadIdx.x & 31;
  const int half = lane >> 4, mn = lane & 15;
  const _Float16* base = P + mn * ld + k0 + half * 8;
  v8h lo = *(const v8h*)(base);
  v8h hi = *(const v8h*)(base + 16);
  v16h f;
#pragma unroll
  for (int e = 0; e < 8; ++e) { f[e] = lo[e]; f[8 + e] = hi[e]; }
  return f;
}

// acc += A(16xK, f16) * W(16xK, f16)^T
__device__ __forceinline__ v8f wmma_16w16(const _Float16* A, int lda,
                                          const _Float16* W, int ldw,
                                          int K, v8f acc) {
  for (int k0 = 0; k0 < K; k0 += 32)
    acc = __builtin_amdgcn_wmma_f32_16x16x32_f16(false, frag_f16(A, lda, k0),
                                                 false, frag_f16(W, ldw, k0),
                                                 (short)0, acc, false, false);
  return acc;
}

// ---------------------------------------------------------------------------
// One-time down-convert f32 -> f16 (weights).
// ---------------------------------------------------------------------------
__global__ void f32_to_f16(const float* __restrict__ src,
                           _Float16* __restrict__ dst, int n) {
  int i = blockIdx.x * 256 + threadIdx.x;
  if (i < n) dst[i] = (_Float16)src[i];
}

// ---------------------------------------------------------------------------
// Embedding lookup (f16 out; consumed only as WMMA A operand):
// X0[(t*B+b)][e] = (x[b][t]==PAD) ? 0 : emb[x[b][t]][e]
// ---------------------------------------------------------------------------
__global__ void embed_kernel(const int* __restrict__ x,
                             const float* __restrict__ emb,
                             _Float16* __restrict__ X0) {
  int row = blockIdx.x;              // t*B + b
  int t = row >> 5, b = row & 31;
  int ch = x[b * T_ENC + t];
  float v = (ch == PAD_TOK) ? 0.f : emb[ch * EMB + threadIdx.x];
  X0[(size_t)row * EMB + threadIdx.x] = (_Float16)v;
}

// ---------------------------------------------------------------------------
// C[M,N](f32) = A[M,K](f16) @ W[N,K](f16)^T + bias[N].  block = 256 (8
// waves), each wave owns one 16x16 tile; grid = (M/16, N/128).
// ---------------------------------------------------------------------------
__global__ __launch_bounds__(256)
void gemm_xwT(const _Float16* __restrict__ A, const _Float16* __restrict__ W,
              const float* __restrict__ bias, float* __restrict__ C,
              int N, int K) {
  const int wave = threadIdx.x >> 5;
  const int lane = threadIdx.x & 31;
  const int half = lane >> 4, mn = lane & 15;
  const int m0 = blockIdx.x * 16;
  const int n0 = (blockIdx.y * 8 + wave) * 16;
  v8f acc;
  float bv = bias ? bias[n0 + mn] : 0.f;
#pragma unroll
  for (int r = 0; r < 8; ++r) acc[r] = bv;
  acc = wmma_16w16(A + (size_t)m0 * K, K, W + (size_t)n0 * K, K, K, acc);
#pragma unroll
  for (int r = 0; r < 8; ++r)
    C[(size_t)(m0 + half * 8 + r) * N + n0 + mn] = acc[r];
}

// ---------------------------------------------------------------------------
// Persistent LSTM recurrent scan over T steps.  blockIdx.x = direction
// (0=fwd, 1=bwd).  Gx = precomputed X@Wih^T + b, shape (T, B, 1024) f32.
// 512 threads = 16 waves; wave j owns hidden columns [16j,16j+16) and keeps
// its c slice in registers (f32).  h lives in LDS as f16 (WMMA consumes f16
// anyway).  Output Hout (T, B, 512) f16 feeds the next layer's GEMM only.
// ---------------------------------------------------------------------------
__global__ __launch_bounds__(512)
void lstm_scan(const float* __restrict__ Gxf, const float* __restrict__ Gxb,
               const _Float16* __restrict__ Whhf,
               const _Float16* __restrict__ Whhb,
               const int* __restrict__ lx, _Float16* __restrict__ Hout) {
  __shared__ alignas(16) _Float16 hS[BN][ENC_H];
  const int dir = blockIdx.x;
  const float* Gx = dir ? Gxb : Gxf;
  const _Float16* Whh = dir ? Whhb : Whhf;
  const int tid = threadIdx.x;
  const int wave = tid >> 5, lane = tid & 31;
  const int half = lane >> 4, mn = lane & 15;
  const int j = wave;  // hidden column tile

  for (int i = tid; i < BN * ENC_H; i += 512) (&hS[0][0])[i] = (_Float16)0.f;
  float c_reg[2][8];
#pragma unroll
  for (int m = 0; m < 2; ++m)
#pragma unroll
    for (int r = 0; r < 8; ++r) c_reg[m][r] = 0.f;

  for (int step = 0; step < T_ENC; ++step) {
    const int t = dir ? (T_ENC - 1 - step) : step;
    __syncthreads();  // previous h writes visible
    v8f g[4][2];
#pragma unroll
    for (int q = 0; q < 4; ++q)
#pragma unroll
      for (int m = 0; m < 2; ++m) {
        const int n0 = q * ENC_H + j * 16;
        const int m0 = m * 16;
        v8f acc;
#pragma unroll
        for (int r = 0; r < 8; ++r)
          acc[r] = Gx[((size_t)t * BN + m0 + half * 8 + r) * 1024 + n0 + mn];
        g[q][m] = wmma_16w16(&hS[m0][0], ENC_H,
                             Whh + (size_t)n0 * ENC_H, ENC_H, ENC_H, acc);
      }
    __syncthreads();  // all h reads done
#pragma unroll
    for (int m = 0; m < 2; ++m)
#pragma unroll
      for (int r = 0; r < 8; ++r) {
        const int rb = m * 16 + half * 8 + r;          // batch row
        const int col = j * 16 + mn;                    // hidden col
        float iv = sigm(g[0][m][r]);
        float fv = sigm(g[1][m][r]);
        float gv = tanhf(g[2][m][r]);
        float ov = sigm(g[3][m][r]);
        float c = fv * c_reg[m][r] + iv * gv;
        float h = ov * tanhf(c);
        bool valid = t < lx[rb];
        float hold = (float)hS[rb][col];
        if (valid) c_reg[m][r] = c;
        hS[rb][col] = (_Float16)(valid ? h : hold);
        Hout[((size_t)t * BN + rb) * (2 * ENC_H) + dir * ENC_H + col] =
            (_Float16)(valid ? h : 0.f);
      }
  }
}

// ---------------------------------------------------------------------------
// Attention phase (device helper used by the decoder).  128 (b,head) pairs
// over 16 waves, 8 pairs each, lane-parallel over t.
// Writes ctx into qctxS[b][128..255] (f32, for logits) and ctxH (f16, for
// the gates1 WMMA); for b==0 writes the weights to out_att.
// Must be called by all 512 threads (contains uniform __syncthreads).
// ---------------------------------------------------------------------------
__device__ void attn_phase(int att_row,
                           const float* __restrict__ Kb,
                           const float* __restrict__ Vb,
                           const int* __restrict__ lx,
                           float (*qctxS)[2 * PROJ],
                           _Float16 (*ctxH)[PROJ],
                           float (*wS)[T_ENC],
                           float* __restrict__ out_att) {
  const int tid = threadIdx.x, wave = tid >> 5, lane = tid & 31;
  for (int i = 0; i < 8; ++i) {
    const int pair = wave * 8 + i;      // 0..127
    const int b = pair >> 2, hd = pair & 3;
    const int L = lx[b];
    float s[8];
    float mx = -1e30f;
#pragma unroll
    for (int k = 0; k < 8; ++k) {
      const int t = lane + 32 * k;
      const float* kp = Kb + ((size_t)t * BN + b) * PROJ + hd * HD;
      const float* qp = &qctxS[b][hd * HD];
      float acc = 0.f;
#pragma unroll
      for (int d = 0; d < HD; ++d) acc += qp[d] * kp[d];
      s[k] = (t < L) ? acc * ATT_SCALE : -1e9f;
      mx = fmaxf(mx, s[k]);
    }
    for (int o = 16; o; o >>= 1) mx = fmaxf(mx, __shfl_xor(mx, o));
    float sum = 0.f;
#pragma unroll
    for (int k = 0; k < 8; ++k) { s[k] = __expf(s[k] - mx); sum += s[k]; }
    for (int o = 16; o; o >>= 1) sum += __shfl_xor(sum, o);
    const float inv = 1.f / sum;
#pragma unroll
    for (int k = 0; k < 8; ++k) {
      const int t = lane + 32 * k;
      const float w = s[k] * inv;
      wS[wave][t] = w;
      if (b == 0)
        out_att[(size_t)att_row * (HEADS * T_ENC) + hd * T_ENC + t] = w;
    }
    __syncthreads();  // wS visible (uniform: all waves loop 8 times)
    float cx = 0.f;
    const float* vp = Vb + (size_t)b * PROJ + hd * HD + lane;
    for (int t = 0; t < T_ENC; ++t) {
      __builtin_prefetch(vp + (size_t)(t + 8) * BN * PROJ, 0, 0);
      cx += wS[wave][t] * vp[(size_t)t * BN * PROJ];
    }
    qctxS[b][PROJ + hd * HD + lane] = cx;
    ctxH[b][hd * HD + lane] = (_Float16)cx;
    __syncthreads();
  }
}

// ---------------------------------------------------------------------------
// Persistent autoregressive decoder: 600 sequential steps in one workgroup.
// 512 threads = 16 waves.  WMMA-only activations (ce, h1, h2, ctx-mirror) in
// f16 LDS; dual-use Q/ctx in f32 LDS; weights pre-converted f16; c in regs.
// ---------------------------------------------------------------------------
__global__ __launch_bounds__(512)
void decoder_kernel(const float* __restrict__ emb, const float* __restrict__ cls_b,
                    const _Float16* __restrict__ Wq, const float* __restrict__ bq,
                    const _Float16* __restrict__ Wih1,
                    const _Float16* __restrict__ Whh1,
                    const float* __restrict__ b1,
                    const _Float16* __restrict__ Wih2,
                    const _Float16* __restrict__ Whh2,
                    const float* __restrict__ b2,
                    const float* __restrict__ init_h1, const float* __restrict__ init_c1,
                    const float* __restrict__ init_h2, const float* __restrict__ init_c2,
                    const float* __restrict__ init_q,
                    const float* __restrict__ Kb, const float* __restrict__ Vb,
                    const int* __restrict__ lx,
                    float* __restrict__ out_pred, float* __restrict__ out_att) {
  __shared__ alignas(16) _Float16 ceS[BN][EMB];    // char emb / init_q staging
  __shared__ alignas(16) _Float16 h1S[BN][DEC_H];
  __shared__ alignas(16) _Float16 h2S[BN][DEC_O];
  __shared__ alignas(16) _Float16 ctxH[BN][PROJ];  // f16 mirror of ctx
  __shared__ float qctxS[BN][2 * PROJ];            // [Q | ctx] == dec_out (f32)
  __shared__ float wS[16][T_ENC];                  // per-wave attention weights
  __shared__ float logitS[BN][32];
  __shared__ int   charS[BN];

  const int tid = threadIdx.x, wave = tid >> 5, lane = tid & 31;
  const int half = lane >> 4, mn = lane & 15;

  for (int i = tid; i < BN * DEC_H; i += 512)
    h1S[i >> 8][i & 255] = (_Float16)init_h1[i & 255];
  for (int i = tid; i < BN * DEC_O; i += 512)
    h2S[i >> 7][i & 127] = (_Float16)init_h2[i & 127];
  for (int i = tid; i < BN * PROJ; i += 512)
    ceS[i >> 7][i & 127] = (_Float16)init_q[i & 127];
  if (tid < BN) charS[tid] = SOS_TOK;
  float c1r[2][8], c2r[2][8];
#pragma unroll
  for (int m = 0; m < 2; ++m)
#pragma unroll
    for (int r = 0; r < 8; ++r) {
      c1r[m][r] = init_c1[wave * 16 + mn];
      c2r[m][r] = (wave < 8) ? init_c2[wave * 16 + mn] : 0.f;
    }
  __syncthreads();

  // Q0 = init_query @ Wq^T + bq  (A rows staged in ceS cols 0..127)
  {
    const int m0 = (wave & 1) * 16, n0 = (wave >> 1) * 16;
    v8f acc;
#pragma unroll
    for (int r = 0; r < 8; ++r) acc[r] = bq[n0 + mn];
    acc = wmma_16w16(&ceS[m0][0], EMB, Wq + (size_t)n0 * DEC_O, DEC_O, DEC_O, acc);
#pragma unroll
    for (int r = 0; r < 8; ++r) qctxS[m0 + half * 8 + r][n0 + mn] = acc[r];
  }
  __syncthreads();
  attn_phase(0, Kb, Vb, lx, qctxS, ctxH, wS, out_att);
  __syncthreads();

  for (int step = 0; step < MAX_STEPS; ++step) {
    // --- char embedding ---
    for (int i = tid; i < BN * EMB; i += 512) {
      const int b = i >> 8, e = i & 255;
      const int ch = charS[b];
      ceS[b][e] = (_Float16)((ch == PAD_TOK) ? 0.f : emb[ch * EMB + e]);
    }
    __syncthreads();

    // --- LSTM1: gates = [ce|ctx]@Wih1^T + h1@Whh1^T + b1 ---
    {
      const int j = wave;
      v8f g[4][2];
#pragma unroll
      for (int q = 0; q < 4; ++q)
#pragma unroll
        for (int m = 0; m < 2; ++m) {
          const int n0 = q * DEC_H + j * 16, m0 = m * 16;
          v8f acc;
#pragma unroll
          for (int r = 0; r < 8; ++r) acc[r] = b1[n0 + mn];
          acc = wmma_16w16(&ceS[m0][0], EMB,
                           Wih1 + (size_t)n0 * 384, 384, EMB, acc);
          acc = wmma_16w16(&ctxH[m0][0], PROJ,
                           Wih1 + (size_t)n0 * 384 + EMB, 384, PROJ, acc);
          acc = wmma_16w16(&h1S[m0][0], DEC_H,
                           Whh1 + (size_t)n0 * DEC_H, DEC_H, DEC_H, acc);
          g[q][m] = acc;
        }
      __syncthreads();
#pragma unroll
      for (int m = 0; m < 2; ++m)
#pragma unroll
        for (int r = 0; r < 8; ++r) {
          const int rb = m * 16 + half * 8 + r;
          float c = sigm(g[1][m][r]) * c1r[m][r] +
                    sigm(g[0][m][r]) * tanhf(g[2][m][r]);
          c1r[m][r] = c;
          h1S[rb][j * 16 + mn] = (_Float16)(sigm(g[3][m][r]) * tanhf(c));
        }
    }
    __syncthreads();

    // --- LSTM2 (waves 0..7): gates = h1@Wih2^T + h2@Whh2^T + b2 ---
    {
      v8f g[4][2];
      if (wave < 8) {
        const int j = wave;
#pragma unroll
        for (int q = 0; q < 4; ++q)
#pragma unroll
          for (int m = 0; m < 2; ++m) {
            const int n0 = q * DEC_O + j * 16, m0 = m * 16;
            v8f acc;
#pragma unroll
            for (int r = 0; r < 8; ++r) acc[r] = b2[n0 + mn];
            acc = wmma_16w16(&h1S[m0][0], DEC_H,
                             Wih2 + (size_t)n0 * DEC_H, DEC_H, DEC_H, acc);
            acc = wmma_16w16(&h2S[m0][0], DEC_O,
                             Whh2 + (size_t)n0 * DEC_O, DEC_O, DEC_O, acc);
            g[q][m] = acc;
          }
      }
      __syncthreads();
      if (wave < 8) {
        const int j = wave;
#pragma unroll
        for (int m = 0; m < 2; ++m)
#pragma unroll
          for (int r = 0; r < 8; ++r) {
            const int rb = m * 16 + half * 8 + r;
            float c = sigm(g[1][m][r]) * c2r[m][r] +
                      sigm(g[0][m][r]) * tanhf(g[2][m][r]);
            c2r[m][r] = c;
            h2S[rb][j * 16 + mn] = (_Float16)(sigm(g[3][m][r]) * tanhf(c));
          }
      }
    }
    __syncthreads();

    // --- Q = h2 @ Wq^T + bq ---
    {
      const int m0 = (wave & 1) * 16, n0 = (wave >> 1) * 16;
      v8f acc;
#pragma unroll
      for (int r = 0; r < 8; ++r) acc[r] = bq[n0 + mn];
      acc = wmma_16w16(&h2S[m0][0], DEC_O, Wq + (size_t)n0 * DEC_O,
                       DEC_O, DEC_O, acc);
#pragma unroll
      for (int r = 0; r < 8; ++r) qctxS[m0 + half * 8 + r][n0 + mn] = acc[r];
    }
    __syncthreads();

    // --- attention ---
    attn_phase(step + 1, Kb, Vb, lx, qctxS, ctxH, wS, out_att);
    __syncthreads();

    // --- weight-tied logits + write out ---
    for (int i = tid; i < BN * VOCAB; i += 512) {
      const int b = i / VOCAB, v = i % VOCAB;
      float acc = cls_b[v];
      const float* ep = emb + v * EMB;
      const float* dp = &qctxS[b][0];
      for (int e = 0; e < 2 * PROJ; ++e) acc += dp[e] * ep[e];
      logitS[b][v] = acc;
      out_pred[(size_t)b * (MAX_STEPS * VOCAB) + step * VOCAB + v] = acc;
    }
    __syncthreads();

    // --- argmax (first max, like jnp.argmax) ---
    if (tid < BN) {
      float best = logitS[tid][0];
      int bi = 0;
      for (int v = 1; v < VOCAB; ++v) {
        float lv = logitS[tid][v];
        if (lv > best) { best = lv; bi = v; }
      }
      charS[tid] = bi;
    }
    __syncthreads();
  }
}

// ---------------------------------------------------------------------------
extern "C" void kernel_launch(void* const* d_in, const int* in_sizes, int n_in,
                              void* d_out, int out_size, void* d_ws, size_t ws_size,
                              hipStream_t stream) {
  (void)out_size; (void)ws_size;
  // Top-level dict may be insertion-ordered (x,lx,params) or pytree-sorted
  // (lx,params...,x): detect from in_sizes[0].
  int ix, ilx, p0;
  if (in_sizes[0] == BN * T_ENC) { ix = 0; ilx = 1; p0 = 2; }
  else                            { ilx = 0; p0 = 1; ix = n_in - 1; }
  const int* x  = (const int*)d_in[ix];
  const int* lx = (const int*)d_in[ilx];
  // params flattened in sorted-key pytree order:
  const float* Whh1  = (const float*)d_in[p0 + 0];
  const float* Whh2  = (const float*)d_in[p0 + 1];
  const float* Wih1  = (const float*)d_in[p0 + 2];
  const float* Wih2  = (const float*)d_in[p0 + 3];
  const float* Wk    = (const float*)d_in[p0 + 4];
  const float* Wq    = (const float*)d_in[p0 + 5];
  const float* Wv    = (const float*)d_in[p0 + 6];
  const float* b1    = (const float*)d_in[p0 + 7];
  const float* b2    = (const float*)d_in[p0 + 8];
  const float* bk    = (const float*)d_in[p0 + 9];
  const float* bq    = (const float*)d_in[p0 + 10];
  const float* bv    = (const float*)d_in[p0 + 11];
  const float* cls_b = (const float*)d_in[p0 + 12];
  const float* emb   = (const float*)d_in[p0 + 13];
  const float* encW[3][2][3];  // [layer][dir 0=f,1=b][0=Whh,1=Wih,2=b]
  for (int l = 0; l < 3; ++l) {
    const int base = p0 + 14 + l * 6;   // keys sorted: 'b' dir before 'f' dir
    encW[l][1][0] = (const float*)d_in[base + 0];
    encW[l][1][1] = (const float*)d_in[base + 1];
    encW[l][1][2] = (const float*)d_in[base + 2];
    encW[l][0][0] = (const float*)d_in[base + 3];
    encW[l][0][1] = (const float*)d_in[base + 4];
    encW[l][0][2] = (const float*)d_in[base + 5];
  }
  const float* init_c1 = (const float*)d_in[p0 + 32];
  const float* init_c2 = (const float*)d_in[p0 + 33];
  const float* init_h1 = (const float*)d_in[p0 + 34];
  const float* init_h2 = (const float*)d_in[p0 + 35];
  const float* init_q  = (const float*)d_in[p0 + 36];

  // workspace layout (all regions 16B-aligned)
  const size_t R = (size_t)T_ENC * BN;            // 8192 rows
  char* wsp = (char*)d_ws;
  _Float16* X0 = (_Float16*)wsp; wsp += R * 256 * sizeof(_Float16);
  _Float16* hA = (_Float16*)wsp; wsp += R * 512 * sizeof(_Float16);
  _Float16* hB = (_Float16*)wsp; wsp += R * 512 * sizeof(_Float16);
  float* Gxf = (float*)wsp;      wsp += R * 1024 * sizeof(float);
  float* Gxb = (float*)wsp;      wsp += R * 1024 * sizeof(float);
  float* Kb  = (float*)wsp;      wsp += R * 128 * sizeof(float);
  float* Vb  = (float*)wsp;      wsp += R * 128 * sizeof(float);
  // f16 weight mirrors (all sizes are multiples of 8 halves -> stay aligned)
  _Float16* h16 = (_Float16*)wsp;
  _Float16* encWih16[3][2];
  _Float16* encWhh16[3][2];
  size_t off = 0;
  for (int l = 0; l < 3; ++l)
    for (int d = 0; d < 2; ++d) {
      encWih16[l][d] = h16 + off; off += (size_t)1024 * (l ? 512 : 256);
      encWhh16[l][d] = h16 + off; off += (size_t)1024 * 256;
    }
  _Float16* Wih1h = h16 + off; off += (size_t)1024 * 384;
  _Float16* Whh1h = h16 + off; off += (size_t)1024 * 256;
  _Float16* Wih2h = h16 + off; off += (size_t)512 * 256;
  _Float16* Whh2h = h16 + off; off += (size_t)512 * 128;
  _Float16* Wqh   = h16 + off; off += (size_t)128 * 128;
  _Float16* Wkh   = h16 + off; off += (size_t)128 * 512;
  _Float16* Wvh   = h16 + off; off += (size_t)128 * 512;

  auto cvt = [&](const float* s, _Float16* d, int n) {
    f32_to_f16<<<(n + 255) / 256, 256, 0, stream>>>(s, d, n);
  };
  for (int l = 0; l < 3; ++l)
    for (int d = 0; d < 2; ++d) {
      cvt(encW[l][d][1], encWih16[l][d], 1024 * (l ? 512 : 256));
      cvt(encW[l][d][0], encWhh16[l][d], 1024 * 256);
    }
  cvt(Wih1, Wih1h, 1024 * 384);
  cvt(Whh1, Whh1h, 1024 * 256);
  cvt(Wih2, Wih2h, 512 * 256);
  cvt(Whh2, Whh2h, 512 * 128);
  cvt(Wq, Wqh, 128 * 128);
  cvt(Wk, Wkh, 128 * 512);
  cvt(Wv, Wvh, 128 * 512);

  embed_kernel<<<R, 256, 0, stream>>>(x, emb, X0);

  const _Float16* in = X0;
  int K = EMB;
  _Float16* out = hA;
  for (int l = 0; l < 3; ++l) {
    dim3 g1(R / 16, 1024 / 128);
    gemm_xwT<<<g1, 256, 0, stream>>>(in, encWih16[l][0], encW[l][0][2], Gxf, 1024, K);
    gemm_xwT<<<g1, 256, 0, stream>>>(in, encWih16[l][1], encW[l][1][2], Gxb, 1024, K);
    lstm_scan<<<2, 512, 0, stream>>>(Gxf, Gxb, encWhh16[l][0], encWhh16[l][1], lx, out);
    in = out;
    K = 2 * ENC_H;
    out = (out == hA) ? hB : hA;
  }
  const _Float16* encH = in;  // (T,B,512) f16 after 3 layers

  gemm_xwT<<<dim3(R / 16, 1), 256, 0, stream>>>(encH, Wkh, bk, Kb, PROJ, 2 * ENC_H);
  gemm_xwT<<<dim3(R / 16, 1), 256, 0, stream>>>(encH, Wvh, bv, Vb, PROJ, 2 * ENC_H);

  float* out_pred = (float*)d_out;                              // (32, 600, 30)
  float* out_att  = out_pred + (size_t)BN * MAX_STEPS * VOCAB;  // (601, 4, 256)
  decoder_kernel<<<1, 512, 0, stream>>>(emb, cls_b, Wqh, bq,
                                        Wih1h, Whh1h, b1, Wih2h, Whh2h, b2,
                                        init_h1, init_c1, init_h2, init_c2,
                                        init_q, Kb, Vb, lx, out_pred, out_att);
}
